// LSTMAutoregressWithPrice_44487271252127
// MI455X (gfx1250) — compile-verified, matrix operand order checked by reading, not run on hardware
//
#include <hip/hip_runtime.h>

// ---------------------------------------------------------------------------
// Problem constants (match reference)
// ---------------------------------------------------------------------------
constexpr int Bn = 64;    // batch
constexpr int Sn = 64;    // encoder steps
constexpr int In = 20;    // input features
constexpr int Hn = 1024;  // hidden
constexpr int Tn = 32;    // decoder steps
constexpr int LPn = 16;   // last-price features

typedef __bf16 bf16;
typedef __attribute__((ext_vector_type(16))) __bf16 v16bf;
typedef __attribute__((ext_vector_type(8)))  __bf16 v8bf;
typedef __attribute__((ext_vector_type(8)))  float  v8f;

// Async global->LDS staging (gfx1250): guarded so the build never regresses.
#if defined(__has_builtin)
#if __has_builtin(__builtin_amdgcn_global_load_async_to_lds_b32) && \
    __has_builtin(__builtin_amdgcn_s_wait_asynccnt)
#define USE_ASYNC_LDS 1
#endif
#endif
#ifndef USE_ASYNC_LDS
#define USE_ASYNC_LDS 0
#endif

union Frag16 { v16bf v; v8bf h[2]; };

// ---------------------------------------------------------------------------
// WMMA fragment load: per-lane, two contiguous 16B chunks along K.
// 16-bit A/B layout: lanes 0-15 hold K {k0..k0+7, k0+16..k0+23},
// lanes 16-31 hold K {k0+8..k0+15, k0+24..k0+31} for row (lane & 15).
// ---------------------------------------------------------------------------
__device__ __forceinline__ v16bf load_frag(const bf16* p, int hofs) {
    Frag16 f;
    f.h[0] = *reinterpret_cast<const v8bf*>(p + hofs);
    f.h[1] = *reinterpret_cast<const v8bf*>(p + 16 + hofs);
    return f.v;
}

// ---------------------------------------------------------------------------
// Core: accumulate a 32x32 output tile (2x2 WMMA 16x16 tiles) of
//   acc += A[m0:m0+32, 0:K] * W[w0:w0+32, 0:K]^T
// A: (rows x lda) bf16 row-major, K contiguous.  W: (rows x ldw) bf16.
// ---------------------------------------------------------------------------
__device__ __forceinline__ void gemm_tile(const bf16* __restrict__ A, int lda, int m0,
                                          const bf16* __restrict__ W, int ldw, int w0,
                                          int K, v8f acc[4]) {
    const int lane = threadIdx.x & 31;
    const int row  = lane & 15;
    const int hofs = (lane >> 4) << 3;
    const bf16* a0 = A + (size_t)(m0 + row) * lda;
    const bf16* a1 = a0 + (size_t)16 * lda;
    const bf16* b0 = W + (size_t)(w0 + row) * ldw;
    const bf16* b1 = b0 + (size_t)16 * ldw;
    for (int k = 0; k < K; k += 32) {
        // locality 3 -> WGP-scope prefetch: pulls the L2-resident weight
        // stream into the near cache ahead of the load clause.
        __builtin_prefetch(b0 + k + 256, 0, 3);
        __builtin_prefetch(b1 + k + 256, 0, 3);
        v16bf fa0 = load_frag(a0 + k, hofs);
        v16bf fa1 = load_frag(a1 + k, hofs);
        v16bf fb0 = load_frag(b0 + k, hofs);
        v16bf fb1 = load_frag(b1 + k, hofs);
        acc[0] = __builtin_amdgcn_wmma_f32_16x16x32_bf16(false, fa0, false, fb0, (short)0, acc[0], false, false);
        acc[1] = __builtin_amdgcn_wmma_f32_16x16x32_bf16(false, fa0, false, fb1, (short)0, acc[1], false, false);
        acc[2] = __builtin_amdgcn_wmma_f32_16x16x32_bf16(false, fa1, false, fb0, (short)0, acc[2], false, false);
        acc[3] = __builtin_amdgcn_wmma_f32_16x16x32_bf16(false, fa1, false, fb1, (short)0, acc[3], false, false);
    }
}

__device__ __forceinline__ float sigm(float x) { return 1.0f / (1.0f + __expf(-x)); }
__device__ __forceinline__ float tanh_fast(float x) { return 1.0f - 2.0f / (__expf(2.0f * x) + 1.0f); }

// ---------------------------------------------------------------------------
// Fused LSTM cell step:
//   gates = X @ Wih^T + Hprev @ Whh^T + bih + bhh   (computed via WMMA)
//   c = sig(f)*c + sig(i)*tanh(g);  h = sig(o)*tanh(c)
// Block: 32-wide hidden strip, all 4 gates, full batch (M=64).
// 8 waves = 2(M) x 4(gate); each wave: 2x2 WMMA tiles (32x32).
// Gates staged in LDS; pointwise update fused in epilogue.
// Bias slices are async-staged global->LDS at entry, overlapped with the
// WMMA K-loop, consumed after s_wait_asynccnt + barrier.
// Grid: H/32 = 32 blocks, 256 threads.
// ---------------------------------------------------------------------------
__global__ void lstm_cell_kernel(const bf16* __restrict__ X, int Kx,
                                 const bf16* __restrict__ Wih,
                                 const bf16* __restrict__ Hprev,
                                 const bf16* __restrict__ Whh,
                                 const float* __restrict__ bih,
                                 const float* __restrict__ bhh,
                                 float* __restrict__ C,
                                 bf16* __restrict__ Hout) {
    __shared__ float lds[64][132];   // 64 batch rows x (4 gates * 32 cols), padded
    __shared__ float biasS[256];     // [0:128)=bih slice, [128:256)=bhh slice (gate-major)

    const int t = threadIdx.x;
    const int wave = t >> 5;
    const int wm = wave >> 2;        // 0..1 : M half
    const int g  = wave & 3;         // gate index (i,f,g,o)
    const int n0 = blockIdx.x * 32;  // hidden strip
    const int wrow = g * Hn + n0;    // weight row base for this gate
    const int m0 = wm * 32;

    // Stage this block's bias slices into LDS, overlapped with the GEMM.
    {
        const float* src = ((t < 128) ? bih : bhh) + ((t & 127) >> 5) * Hn + n0 + (t & 31);
#if USE_ASYNC_LDS
        __builtin_amdgcn_global_load_async_to_lds_b32(
            (__attribute__((address_space(1))) int*)src,
            (__attribute__((address_space(3))) int*)&biasS[t], 0, 0);
#else
        biasS[t] = *src;
#endif
    }

    v8f acc[4] = {};
    gemm_tile(X, Kx, m0, Wih, Kx, wrow, Kx, acc);       // input contribution
    gemm_tile(Hprev, Hn, m0, Whh, Hn, wrow, Hn, acc);   // recurrent contribution

#if USE_ASYNC_LDS
    __builtin_amdgcn_s_wait_asynccnt(0);
#endif

    // Scatter accumulators to LDS. C/D layout: VGPR r -> (M = r + (lane<16?0:8), N = lane&15)
    const int lane = t & 31;
    const int tn = lane & 15;
    const int tm = (lane >> 4) << 3;
    for (int mi = 0; mi < 2; ++mi) {
        for (int ni = 0; ni < 2; ++ni) {
            v8f a = acc[mi * 2 + ni];
            int mb = m0 + mi * 16 + tm;
            int cb = g * 32 + ni * 16 + tn;
            #pragma unroll
            for (int r = 0; r < 8; ++r) lds[mb + r][cb] = a[r];
        }
    }
    __syncthreads();

    // Pointwise LSTM update: 64x32 elements, 8 per thread.
    for (int idx = t; idx < 64 * 32; idx += 256) {
        int m = idx >> 5;
        int n = idx & 31;
        int hidx = n0 + n;
        float gi = lds[m][n]      + biasS[n]      + biasS[128 + n];
        float gf = lds[m][32 + n] + biasS[32 + n] + biasS[160 + n];
        float gg = lds[m][64 + n] + biasS[64 + n] + biasS[192 + n];
        float go = lds[m][96 + n] + biasS[96 + n] + biasS[224 + n];
        float c_old = C[m * Hn + hidx];
        float c_new = sigm(gf) * c_old + sigm(gi) * tanh_fast(gg);
        float h_new = sigm(go) * tanh_fast(c_new);
        C[m * Hn + hidx] = c_new;
        Hout[m * Hn + hidx] = (bf16)h_new;
    }
}

// ---------------------------------------------------------------------------
// FC1: out = relu(concat(h1, lp) @ W1^T + b1)  -> (64, 1024) f32
// Two accumulated WMMA GEMMs: h part (K=1024) + zero-padded lp part (K=32).
// Grid: 1024/(4*32) = 8 blocks, 256 threads (2x4 waves, 32x32 tile/wave).
// ---------------------------------------------------------------------------
__global__ void fc1_kernel(const bf16* __restrict__ H1,
                           const bf16* __restrict__ LPb,
                           const bf16* __restrict__ W1a,
                           const bf16* __restrict__ W1b,
                           const float* __restrict__ b1,
                           float* __restrict__ out) {
    const int wave = threadIdx.x >> 5;
    const int wm = wave >> 2;
    const int wn = wave & 3;
    const int n0 = blockIdx.x * 128 + wn * 32;
    const int m0 = wm * 32;

    v8f acc[4] = {};
    gemm_tile(H1, Hn, m0, W1a, Hn, n0, Hn, acc);
    gemm_tile(LPb, 32, m0, W1b, 32, n0, 32, acc);

    const int lane = threadIdx.x & 31;
    const int tn = lane & 15;
    const int tm = (lane >> 4) << 3;
    for (int mi = 0; mi < 2; ++mi) {
        for (int ni = 0; ni < 2; ++ni) {
            v8f a = acc[mi * 2 + ni];
            int nb = n0 + ni * 16 + tn;
            int mb = m0 + mi * 16 + tm;
            float bv = b1[nb];
            #pragma unroll
            for (int r = 0; r < 8; ++r) {
                float v = a[r] + bv;
                out[(mb + r) * Hn + nb] = v > 0.0f ? v : 0.0f;
            }
        }
    }
}

// ---------------------------------------------------------------------------
// FC2 + output + autoregressive state update (tiny: 64x20x1024).
// One wave per (batch, output) pair; 1280 waves total.
// ---------------------------------------------------------------------------
__global__ void fc2_kernel(const float* __restrict__ fc,
                           const float* __restrict__ W2,
                           const float* __restrict__ b2,
                           float* __restrict__ lp,
                           bf16* __restrict__ cur,
                           float* __restrict__ out, int t) {
    int w = (blockIdx.x * blockDim.x + threadIdx.x) >> 5;
    int lane = threadIdx.x & 31;
    if (w >= Bn * In) return;
    int b = w / In, n = w % In;
    const float* fr = fc + b * Hn;
    const float* wr = W2 + n * Hn;
    float s = 0.0f;
    for (int k = lane; k < Hn; k += 32) s += fr[k] * wr[k];
    #pragma unroll
    for (int off = 16; off > 0; off >>= 1) s += __shfl_down(s, off, 32);
    if (lane == 0) {
        float pred = s + b2[n];
        out[b * (Tn * In) + t * In + n] = pred;
        cur[b * 32 + n] = (bf16)pred;     // next decoder input (pad cols stay 0)
        int q = n % 5;
        if (q < 4) lp[b * LPn + (n / 5) * 4 + q] += pred * 0.01f;
    }
}

// ---------------------------------------------------------------------------
// Small utility kernels
// ---------------------------------------------------------------------------
__global__ void cvt_pad_kernel(bf16* __restrict__ dst, int dstld,
                               const float* __restrict__ src, int srcld,
                               int valid, int rows) {
    int n = rows * dstld;
    for (int i = blockIdx.x * blockDim.x + threadIdx.x; i < n; i += gridDim.x * blockDim.x) {
        int r = i / dstld, c = i - r * dstld;
        dst[i] = (c < valid) ? (bf16)src[(size_t)r * srcld + c] : (bf16)0.0f;
    }
}

__global__ void zero_f32_kernel(float* p, int n) {
    for (int i = blockIdx.x * blockDim.x + threadIdx.x; i < n; i += gridDim.x * blockDim.x) p[i] = 0.0f;
}
__global__ void zero_bf16_kernel(bf16* p, int n) {
    for (int i = blockIdx.x * blockDim.x + threadIdx.x; i < n; i += gridDim.x * blockDim.x) p[i] = (bf16)0.0f;
}
__global__ void copy_f32_kernel(float* d, const float* s, int n) {
    for (int i = blockIdx.x * blockDim.x + threadIdx.x; i < n; i += gridDim.x * blockDim.x) d[i] = s[i];
}

// x[:, t, :] -> cur_bf (64 x 32, cols 20..31 already zero)
__global__ void prep_x_kernel(bf16* __restrict__ cur, const float* __restrict__ x, int t) {
    int i = blockIdx.x * blockDim.x + threadIdx.x;
    if (i < Bn * In) {
        int b = i / In, c = i - b * In;
        cur[b * 32 + c] = (bf16)x[b * (Sn * In) + t * In + c];
    }
}

// lp (f32 64x16) -> lp_bf (64x32, pad zero)
__global__ void prep_lp_kernel(bf16* __restrict__ lpb, const float* __restrict__ lp) {
    int i = blockIdx.x * blockDim.x + threadIdx.x;
    if (i < Bn * LPn) {
        int b = i >> 4, c = i & 15;
        lpb[b * 32 + c] = (bf16)lp[i];
    }
}

// ---------------------------------------------------------------------------
// Host orchestration (graph-capture safe: launches only, all on `stream`)
// ---------------------------------------------------------------------------
extern "C" void kernel_launch(void* const* d_in, const int* in_sizes, int n_in,
                              void* d_out, int out_size, void* d_ws, size_t ws_size,
                              hipStream_t stream) {
    (void)in_sizes; (void)n_in; (void)out_size; (void)ws_size;

    const float* x    = (const float*)d_in[0];
    const float* lp0  = (const float*)d_in[1];
    const float* Wih0 = (const float*)d_in[2];
    const float* Whh0 = (const float*)d_in[3];
    const float* bih0 = (const float*)d_in[4];
    const float* bhh0 = (const float*)d_in[5];
    const float* Wih1 = (const float*)d_in[6];
    const float* Whh1 = (const float*)d_in[7];
    const float* bih1 = (const float*)d_in[8];
    const float* bhh1 = (const float*)d_in[9];
    const float* W1   = (const float*)d_in[10];
    const float* b1   = (const float*)d_in[11];
    const float* W2   = (const float*)d_in[12];
    const float* b2   = (const float*)d_in[13];
    float* out = (float*)d_out;

    // Workspace carve (~28.5 MB)
    char* ws = (char*)d_ws;
    size_t o = 0;
    auto alloc = [&](size_t bytes) -> char* {
        char* p = ws + o;
        o += (bytes + 255) & ~(size_t)255;
        return p;
    };
    bf16* Whh0b = (bf16*)alloc((size_t)4 * Hn * Hn * 2);
    bf16* Whh1b = (bf16*)alloc((size_t)4 * Hn * Hn * 2);
    bf16* Wih1b = (bf16*)alloc((size_t)4 * Hn * Hn * 2);
    bf16* Wih0b = (bf16*)alloc((size_t)4 * Hn * 32 * 2);
    bf16* W1ab  = (bf16*)alloc((size_t)Hn * Hn * 2);
    bf16* W1bb  = (bf16*)alloc((size_t)Hn * 32 * 2);
    bf16* h0b[2] = { (bf16*)alloc((size_t)Bn * Hn * 2), (bf16*)alloc((size_t)Bn * Hn * 2) };
    bf16* h1b[2] = { (bf16*)alloc((size_t)Bn * Hn * 2), (bf16*)alloc((size_t)Bn * Hn * 2) };
    float* c0    = (float*)alloc((size_t)Bn * Hn * 4);
    float* c1    = (float*)alloc((size_t)Bn * Hn * 4);
    bf16* curbf  = (bf16*)alloc((size_t)Bn * 32 * 2);
    bf16* lpbf   = (bf16*)alloc((size_t)Bn * 32 * 2);
    float* lp    = (float*)alloc((size_t)Bn * LPn * 4);
    float* fcrel = (float*)alloc((size_t)Bn * Hn * 4);

    auto cvt = [&](bf16* dst, int dstld, const float* src, int srcld, int valid, int rows) {
        int n = rows * dstld;
        cvt_pad_kernel<<<(n + 255) / 256, 256, 0, stream>>>(dst, dstld, src, srcld, valid, rows);
    };

    // Weight conversion f32 -> bf16 (padded where needed)
    cvt(Whh0b, Hn, Whh0, Hn, Hn, 4 * Hn);
    cvt(Whh1b, Hn, Whh1, Hn, Hn, 4 * Hn);
    cvt(Wih1b, Hn, Wih1, Hn, Hn, 4 * Hn);
    cvt(Wih0b, 32, Wih0, In, In, 4 * Hn);            // K padded 20 -> 32
    cvt(W1ab,  Hn, W1, Hn + LPn, Hn, Hn);            // W1[:, :1024]
    cvt(W1bb,  32, W1 + Hn, Hn + LPn, LPn, Hn);      // W1[:, 1024:1040], K padded -> 32

    // State init (every call; harness poisons ws once, graph re-runs this)
    zero_f32_kernel<<<256, 256, 0, stream>>>(c0, Bn * Hn);
    zero_f32_kernel<<<256, 256, 0, stream>>>(c1, Bn * Hn);
    zero_bf16_kernel<<<256, 256, 0, stream>>>(h0b[0], Bn * Hn);
    zero_bf16_kernel<<<256, 256, 0, stream>>>(h0b[1], Bn * Hn);
    zero_bf16_kernel<<<256, 256, 0, stream>>>(h1b[0], Bn * Hn);
    zero_bf16_kernel<<<256, 256, 0, stream>>>(h1b[1], Bn * Hn);
    zero_bf16_kernel<<<8, 256, 0, stream>>>(curbf, Bn * 32);
    zero_bf16_kernel<<<8, 256, 0, stream>>>(lpbf, Bn * 32);
    copy_f32_kernel<<<4, 256, 0, stream>>>(lp, lp0, Bn * LPn);

    // 64 encoder + 32 decoder steps; h buffers ping-pong on step parity.
    for (int g = 0; g < Sn + Tn; ++g) {
        const bool enc = g < Sn;
        const int p = g & 1;
        bf16* h0_old = h0b[p];
        bf16* h0_new = h0b[1 - p];
        bf16* h1_old = h1b[p];
        bf16* h1_new = h1b[1 - p];

        if (enc) prep_x_kernel<<<5, 256, 0, stream>>>(curbf, x, g);

        // layer 0: input = curbf (K=32 padded), layer 1: input = new h0 (K=1024)
        lstm_cell_kernel<<<Hn / 32, 256, 0, stream>>>(curbf, 32, Wih0b, h0_old, Whh0b,
                                                      bih0, bhh0, c0, h0_new);
        lstm_cell_kernel<<<Hn / 32, 256, 0, stream>>>(h0_new, Hn, Wih1b, h1_old, Whh1b,
                                                      bih1, bhh1, c1, h1_new);
        if (!enc) {
            prep_lp_kernel<<<4, 256, 0, stream>>>(lpbf, lp);
            fc1_kernel<<<Hn / 128, 256, 0, stream>>>(h1_new, lpbf, W1ab, W1bb, b1, fcrel);
            fc2_kernel<<<(Bn * In * 32 + 255) / 256, 256, 0, stream>>>(fcrel, W2, b2, lp,
                                                                       curbf, out, g - Sn);
        }
    }
}
